// ContrastiveNoAugLoss_27178553049632
// MI455X (gfx1250) — matrix operand order repaired and checked by reading, not compiled, not run on hardware
//
#include <hip/hip_runtime.h>
#include <hip/hip_bf16.h>

typedef __attribute__((ext_vector_type(16))) _Float16 v16h;
typedef __attribute__((ext_vector_type(8)))  _Float16 v8h;
typedef __attribute__((ext_vector_type(8)))  float    v8f;

#define WIN 7
#define OUTW 58
#define NIMG 64
#define IMGW 64
#define NPOS (OUTW * OUTW)

__device__ __forceinline__ v8f wmma_f16(v16h a, v16h b, v8f c) {
    // D = A(16x32 f16) * B(32x16 f16) + C(16x16 f32)
    return __builtin_amdgcn_wmma_f32_16x16x32_f16(
        /*neg_a=*/false, a, /*neg_b=*/false, b,
        /*c_mod=*/(short)0, c, /*reuse_a=*/false, /*reuse_b=*/false);
}

// gfx1250 async bulk copy: global -> LDS, 16B per lane, ASYNCcnt-tracked, no VGPR data.
__device__ __forceinline__ void async_g2l_b128(void* lds_dst, const void* gsrc) {
    unsigned lds = (unsigned)(unsigned long long)lds_dst;   // low 32 bits of generic = LDS offset
    unsigned long long g = (unsigned long long)gsrc;
    asm volatile("global_load_async_to_lds_b128 %0, %1, off"
                 :: "v"(lds), "v"(g) : "memory");
}
__device__ __forceinline__ void wait_async_zero() {
    asm volatile("s_wait_asynccnt 0x0" ::: "memory");
}

__device__ __forceinline__ float ssim_term(float g, float mui, float muj, float vi, float vj) {
    const float C1 = 4.0e-4f;            // (0.01*2)^2
    const float C2 = 3.6e-3f;            // (0.03*2)^2
    const float COVN = 49.0f / 48.0f;
    const float INV49 = 1.0f / 49.0f;
    float mxy = g * INV49;
    float a   = mui * muj;
    float cov = COVN * (mxy - a);
    float num = (2.0f * a + C1) * (2.0f * cov + C2);
    float den = (mui * mui + muj * muj + C1) * (vi + vj + C2);
    return num * __builtin_amdgcn_rcpf(den);
}

// ================= K0: z cosine Gram + zero the SSIM accumulator =================
__global__ void k_zcos(const float* __restrict__ z, float* __restrict__ zcos,
                       float* __restrict__ accum) {
    __shared__ float sinv[64];
    int tid = threadIdx.x;
    if (tid < 64) {
        float s = 0.0f;
        const float* row = z + tid * 256;
        for (int k = 0; k < 256; ++k) { float v = row[k]; s += v * v; }
        float n = sqrtf(s);
        sinv[tid] = 1.0f / fmaxf(n, 1e-12f);
    }
    __syncthreads();
    for (int o = tid; o < 4096; o += 256) {
        int i = o >> 6, j = o & 63;
        const float* ri = z + i * 256;
        const float* rj = z + j * 256;
        float d = 0.0f;
        for (int k = 0; k < 256; ++k) d = fmaf(ri[k], rj[k], d);
        zcos[o]  = d * sinv[i] * sinv[j];
        accum[o] = 0.0f;
    }
}

// ================= K1: per-image mu/var (7x7 box) + f16 repack [c][h][img][w] =====
__global__ void k_prep(const float* __restrict__ x, _Float16* __restrict__ xh,
                       float* __restrict__ mu, float* __restrict__ var) {
    __shared__ float img[4096];
    int bx = blockIdx.x;
    int i = bx / 3, c = bx % 3;
    int tid = threadIdx.x;
    const float* src = x + ((size_t)i * 3 + c) * 4096;
    for (int p = tid; p < 4096; p += 256) img[p] = src[p];
    __syncthreads();
    // f16 repack: xh[c][h][i][w]
    for (int p = tid; p < 4096; p += 256) {
        int h = p >> 6, w = p & 63;
        xh[(((size_t)c * 64 + h) * NIMG + i) * IMGW + w] = (_Float16)img[p];
    }
    // mu/var: layout [c][h(58)][img][w(58)]
    for (int pos = tid; pos < NPOS; pos += 256) {
        int h = pos / OUTW, w = pos - (pos / OUTW) * OUTW;
        float s = 0.0f, sq = 0.0f;
        for (int dy = 0; dy < WIN; ++dy) {
#pragma unroll
            for (int dx = 0; dx < WIN; ++dx) {
                float v = img[(h + dy) * 64 + (w + dx)];
                s  += v;
                sq = fmaf(v, v, sq);
            }
        }
        float m  = s * (1.0f / 49.0f);
        float vr = (49.0f / 48.0f) * (sq * (1.0f / 49.0f) - m * m);
        size_t o = (((size_t)c * OUTW + h) * NIMG + i) * OUTW + w;
        mu[o]  = m;
        var[o] = vr;
    }
}

// ================= K2: all-pairs Gram via WMMA + SSIM accumulation ================
// grid = 3 channels * 58 output rows; block = 256 (8 wave32)
__global__ void __launch_bounds__(256) k_main(const _Float16* __restrict__ xh,
                                              const float* __restrict__ mu,
                                              const float* __restrict__ var,
                                              float* __restrict__ accum) {
    extern __shared__ char smem[];
    _Float16* ximg = (_Float16*)smem;                         // 7*64*64 halves = 57344 B
    _Float16* pbuf = (_Float16*)(smem + 57344);               // ping-pong im2col: 2*4096 halves
    float*    smu  = (float*)(smem + 57344 + 16384);          // 64*58 f32 = 14848 B
    float*    svar = smu + NIMG * OUTW;                       // 64*58 f32 = 14848 B

    int bx = blockIdx.x;
    int c = bx % 3, h = bx / 3;
    int tid = threadIdx.x;

    const char* grow = (const char*)(xh + ((size_t)c * 64 + h) * (NIMG * IMGW));
    const char* gmu  = (const char*)(mu  + ((size_t)c * OUTW + h) * (NIMG * OUTW));
    const char* gvar = (const char*)(var + ((size_t)c * OUTW + h) * (NIMG * OUTW));

    // ---- async bulk staging: global -> LDS, no VGPR round trip (ASYNCcnt) ----
    for (int idx = tid; idx < (7 * NIMG * IMGW * 2) / 16; idx += 256)      // 3584 x b128
        async_g2l_b128((char*)ximg + idx * 16, grow + idx * 16);
    for (int idx = tid; idx < (NIMG * OUTW * 4) / 16; idx += 256) {        // 928 x b128 each
        async_g2l_b128((char*)smu  + idx * 16, gmu  + idx * 16);
        async_g2l_b128((char*)svar + idx * 16, gvar + idx * 16);
    }
    wait_async_zero();
    __syncthreads();

    int wave = tid >> 5, lane = tid & 31;
    int mi   = wave >> 1;           // row tile 0..3
    int njb  = (wave & 1) * 2;      // this wave owns col tiles njb, njb+1
    int half = lane >> 4;           // C/D layout: M = 8*half + v, N = lane&15
    int nr   = lane & 15;
    int img  = tid & 63;            // im2col build: image owned by this thread
    int q    = tid >> 6;            // K-quarter (16 halves) owned by this thread

    v8f sacc0 = {};
    v8f sacc1 = {};

    for (int w = 0; w < OUTW; ++w) {
        _Float16* buf = pbuf + (w & 1) * (NIMG * 64);

        // ---- cooperative im2col: each patch element gathered ONCE per WG ----
        // patch row layout: buf[img*64 + k], k = dy*7+dx (identity perm), zero pad k>=49
        {
            v16h vals;
#pragma unroll
            for (int e = 0; e < 16; ++e) {
                int k = q * 16 + e;
                _Float16 v = (_Float16)0.0f;
                if (k < 49) {
                    int dy = k / 7;
                    int dx = k - dy * 7;
                    v = ximg[(dy * NIMG + img) * IMGW + (w + dx)];
                }
                vals[e] = v;
            }
            *(v16h*)(buf + img * 64 + q * 16) = vals;         // aligned 32B store
        }
        __syncthreads();

        // ---- fragment loads: contiguous k order, same in A and B, fully aligned ----
        // A (16x32, lane L): M = L&15, K = 32s + 16*(L>>4) + e
        const _Float16* arow = buf + (mi * 16 + nr) * 64;
        int akb = 16 * half;
        v16h a0 = *(const v16h*)(arow + akb);
        v16h a1 = *(const v16h*)(arow + 32 + akb);
        // B (32x16, lane L): N = L&15, K = 32s + 16*(L>>4) + e  -> one 32B load
        const _Float16* brow0 = buf + (njb * 16 + nr) * 64;
        const _Float16* brow1 = brow0 + 16 * 64;
        int bkb = 16 * half;
        v16h b00 = *(const v16h*)(brow0 + bkb);
        v16h b01 = *(const v16h*)(brow0 + 32 + bkb);
        v16h b10 = *(const v16h*)(brow1 + bkb);
        v16h b11 = *(const v16h*)(brow1 + 32 + bkb);

        // alternate the two accumulation chains to hide the XDL D->C RAW hazard
        v8f g0 = {};
        v8f g1 = {};
        g0 = wmma_f16(a0, b00, g0);
        g1 = wmma_f16(a0, b10, g1);
        g0 = wmma_f16(a1, b01, g0);
        g1 = wmma_f16(a1, b11, g1);

        int j0 = njb * 16 + nr;
        int j1 = (njb + 1) * 16 + nr;
        float muj0 = smu[j0 * OUTW + w], varj0 = svar[j0 * OUTW + w];
        float muj1 = smu[j1 * OUTW + w], varj1 = svar[j1 * OUTW + w];

#pragma unroll
        for (int vv = 0; vv < 8; ++vv) {
            int M = mi * 16 + half * 8 + vv;
            float mui  = smu[M * OUTW + w];
            float vari = svar[M * OUTW + w];
            sacc0[vv] += ssim_term(g0[vv], mui, muj0, vari, varj0);
            sacc1[vv] += ssim_term(g1[vv], mui, muj1, vari, varj1);
        }
    }

#pragma unroll
    for (int vv = 0; vv < 8; ++vv) {
        int M = mi * 16 + half * 8 + vv;
        unsafeAtomicAdd(&accum[M * 64 + (njb * 16 + nr)],       sacc0[vv]);
        unsafeAtomicAdd(&accum[M * 64 + ((njb + 1) * 16 + nr)], sacc1[vv]);
    }
}

// ================= K3: loss = mean((accum/(3*58*58) - zcos)^2) ===================
__global__ void k_final(const float* __restrict__ zcos, const float* __restrict__ accum,
                        float* __restrict__ out) {
    __shared__ float red[256];
    int tid = threadIdx.x;
    const float scale = 1.0f / (3.0f * (float)NPOS);
    float s = 0.0f;
    for (int o = tid; o < 4096; o += 256) {
        float d = accum[o] * scale - zcos[o];
        s = fmaf(d, d, s);
    }
    red[tid] = s;
    __syncthreads();
    for (int k = 128; k > 0; k >>= 1) {
        if (tid < k) red[tid] += red[tid + k];
        __syncthreads();
    }
    if (tid == 0) out[0] = red[0] * (1.0f / 4096.0f);
}

extern "C" void kernel_launch(void* const* d_in, const int* in_sizes, int n_in,
                              void* d_out, int out_size, void* d_ws, size_t ws_size,
                              hipStream_t stream) {
    (void)in_sizes; (void)n_in; (void)out_size; (void)ws_size;
    const float* z = (const float*)d_in[0];   // [64,256]
    const float* x = (const float*)d_in[1];   // [64,3,64,64]
    float* out = (float*)d_out;

    // workspace carve-up (~6.5 MB total)
    float* zcos  = (float*)d_ws;                        // 4096
    float* accum = zcos + 4096;                         // 4096
    float* mu    = accum + 4096;                        // 3*58*64*58 = 645888
    float* var   = mu + 3 * OUTW * NIMG * OUTW;         // 645888
    _Float16* xh = (_Float16*)(var + 3 * OUTW * NIMG * OUTW); // 3*64*64*64 halves

    k_zcos<<<1, 256, 0, stream>>>(z, zcos, accum);
    k_prep<<<192, 256, 0, stream>>>(x, xh, mu, var);

    const int lds_bytes = 57344 + 16384 + 2 * NIMG * OUTW * 4; // 103424 B < 320KB/WGP
    k_main<<<3 * OUTW, 256, lds_bytes, stream>>>(xh, mu, var, accum);

    k_final<<<1, 256, 0, stream>>>(zcos, accum, out);
}
